// TempPredictor_42460046688920
// MI455X (gfx1250) — matrix operand order repaired and checked by reading, not compiled, not run on hardware
//
#include <hip/hip_runtime.h>

// ---------------------------------------------------------------------------
// GRU (H=32, I=1, T=1024, B=4096) + final FC, CDNA5 (gfx1250) WMMA version.
// One wave (32 lanes) owns 16 batch rows. Per timestep:
//   gates[16,96] = [h,x,1] x [w_hh^T ; w_ih^T ; bias]  via v_wmma_f32_16x16x32_f16
// x is streamed into LDS with the Tensor Data Mover (tensor_load_to_lds),
// double buffered, waited with s_wait_tensorcnt.  All 8 accumulator tiles are
// kept live so the scheduler can interleave WMMAs and hide D->VALU hazards.
// ---------------------------------------------------------------------------

typedef __attribute__((ext_vector_type(16))) _Float16 v16h;
typedef __attribute__((ext_vector_type(8)))  _Float16 v8h;
typedef __attribute__((ext_vector_type(8)))  float    v8f;
typedef __attribute__((ext_vector_type(4)))  unsigned u32x4;
typedef __attribute__((ext_vector_type(8)))  int      i32x8;
typedef __attribute__((ext_vector_type(4)))  int      i32x4;

#ifndef GRU_T
#define GRU_T 1024
#endif
#define GRU_B 4096
#define TCHUNK 64            // timesteps per TDM tile
#define XSTRIDE 65           // 64 dwords + 1 dword TDM pad -> bank spread
#define HSTRIDE 40           // h row stride in halves (16B-aligned, bank spread)

#if defined(__has_builtin)
#if __has_builtin(__builtin_amdgcn_tensor_load_to_lds) && \
    __has_builtin(__builtin_amdgcn_s_wait_tensorcnt)
#define USE_TDM 1
#endif
#endif

__device__ __forceinline__ float fsig(float x) {
  return __builtin_amdgcn_rcpf(1.0f + __expf(-x));
}
__device__ __forceinline__ float ftanh(float x) {
  float e = __expf(-2.0f * x);
  return (1.0f - e) * __builtin_amdgcn_rcpf(1.0f + e);
}

#if defined(USE_TDM)
// 2D TDM: tile = TCHUNK x 16 rows of f32, row stride = T elements in memory,
// pad 1 dword after every 64 dwords written to LDS (row stride -> 65 dwords).
__device__ __forceinline__ void tdm_load_x(const float* gptr, unsigned lds_off) {
  unsigned long long ga = (unsigned long long)(__UINTPTR_TYPE__)gptr;
  u32x4 g0;
  g0[0] = 1u;                                          // count=1 (valid user D#)
  g0[1] = lds_off;                                     // lds_addr (bytes)
  g0[2] = (unsigned)(ga & 0xFFFFFFFFu);                // global_addr[31:0]
  g0[3] = (unsigned)((ga >> 32) & 0x1FFFFFFu)          // global_addr[56:32]
        | (2u << 30);                                  // type = 2 ("image")
  i32x8 g1;
  g1[0] = (2 << 16)        // data_size = 4B
        | (1 << 20)        // pad_enable
        | (5 << 22)        // pad_interval: 64 dwords
        | (0 << 25);       // pad_amount: 1 dword
  g1[1] = (TCHUNK << 16);  // tensor_dim0[15:0] in [31:16]
  g1[2] = (16 << 16);      // tensor_dim0 hi = 0, tensor_dim1[15:0] = 16
  g1[3] = (TCHUNK << 16);  // tensor_dim1 hi = 0, tile_dim0 = TCHUNK
  g1[4] = 16;              // tile_dim1 = 16, tile_dim2 = 0
  g1[5] = GRU_T;           // tensor_dim0_stride lo32 = T elements
  g1[6] = 0;
  g1[7] = 0;
  i32x4 gz4; gz4[0] = 0; gz4[1] = 0; gz4[2] = 0; gz4[3] = 0;
  i32x8 gz8;
#pragma unroll
  for (int i = 0; i < 8; ++i) gz8[i] = 0;
  // clang-23 / therock-10.0 6-arg form (groups 2/3 + extra group unused for 2D).
  __builtin_amdgcn_tensor_load_to_lds(g0, g1, gz4, gz4, gz8, 0);
}
#endif

#define WMMA(A, Bm, C) \
  __builtin_amdgcn_wmma_f32_16x16x32_f16(false, (A), false, (Bm), (short)0, (C), false, false)

__global__ __launch_bounds__(32) void gru_wmma_kernel(
    const float* __restrict__ x,     // [B, T, 1]
    const float* __restrict__ w_ih,  // [96, 1]
    const float* __restrict__ w_hh,  // [96, 32]
    const float* __restrict__ b_ih,  // [96]
    const float* __restrict__ b_hh,  // [96]
    const float* __restrict__ fc_w,  // [1, 32]
    const float* __restrict__ fc_b,  // [1]
    float* __restrict__ out)         // [B, 1]
{
  __shared__ __align__(16) float    xlds[2][16 * XSTRIDE];
  __shared__ __align__(16) _Float16 hlds[16 * HSTRIDE];

  const int lane = threadIdx.x;          // 0..31 (wave32)
  const int col  = lane & 15;            // tile column / batch-row id
  const int hi   = lane >> 4;            // lane half
  const int bbase = blockIdx.x * 16;     // 16 batch rows per wave

  // ---- Resident B-operand tiles (32x16 f16). Layout: lane l, slot s holds
  // element (K = s + 16*(l>>4), N = l&15).  whh[j] = w_hh^T cols 16j..16j+15.
  v16h whh[6], wib[6];
#pragma unroll
  for (int j = 0; j < 6; ++j) {
    const float* wrow = w_hh + (16 * j + col) * 32 + 16 * hi;  // w_hh[16j+N][K..]
    v16h w;
#pragma unroll
    for (int s = 0; s < 16; ++s) w[s] = (_Float16)wrow[s];
    whh[j] = w;
    // wib: virtual 32x16 with row0 = w_ih, row1 = bias, rows 2..31 = 0.
    v16h t;
#pragma unroll
    for (int s = 0; s < 16; ++s) t[s] = (_Float16)0.0f;
    if (hi == 0) {
      t[0] = (_Float16)w_ih[16 * j + col];
      float bias = b_ih[16 * j + col];
      if (j < 4) bias += b_hh[16 * j + col];   // fuse b_hh for r/z gates
      t[1] = (_Float16)bias;
    }
    wib[j] = t;
  }
  const float bh0 = b_hh[64 + col];  // b_hn, cols 0..15
  const float bh1 = b_hh[80 + col];  // b_hn, cols 16..31

  // ---- Loop-carried state.
  v16h hA;                                   // h in A-layout (16x32 f16)
#pragma unroll
  for (int s = 0; s < 16; ++s) hA[s] = (_Float16)0.0f;
  v8f hD0, hD1, zero;                        // h in D-layout (f32), cols 0-15 / 16-31
#pragma unroll
  for (int v = 0; v < 8; ++v) { hD0[v] = 0.0f; hD1[v] = 0.0f; zero[v] = 0.0f; }
  v16h ax;                                   // A2 = [x, 1, 0...] (rows = batch)
#pragma unroll
  for (int s = 0; s < 16; ++s) ax[s] = (_Float16)0.0f;
  if (hi == 0) ax[1] = (_Float16)1.0f;

  const float* xbase = x + (__SIZE_TYPE__)bbase * GRU_T;

#if defined(USE_TDM)
  tdm_load_x(xbase, (unsigned)(__UINTPTR_TYPE__)&xlds[0][0]);  // pre-issue chunk 0
#endif

  for (int t0 = 0; t0 < GRU_T; t0 += TCHUNK) {
    const int buf = (t0 / TCHUNK) & 1;
    const float* xrow = &xlds[buf][col * XSTRIDE];
#if defined(USE_TDM)
    __builtin_amdgcn_s_wait_tensorcnt(0);    // current chunk resident in LDS
    // Escape the LDS pointer + full memory clobber: loads below must re-read
    // LDS (TDM wrote it behind the compiler's back) and cannot hoist above
    // the tensorcnt wait.  Plain ds_load_b32 afterwards (NOT volatile: a
    // volatile shared access lowers to a flat SYS-scope load).
    __asm__ __volatile__("" : : "v"(xrow) : "memory");
    if (t0 + TCHUNK < GRU_T)                 // prefetch next chunk (double buffer)
      tdm_load_x(xbase + t0 + TCHUNK,
                 (unsigned)(__UINTPTR_TYPE__)&xlds[buf ^ 1][0]);
#else
    for (int idx = lane; idx < 16 * TCHUNK; idx += 32) {
      int r = idx / TCHUNK, c = idx % TCHUNK;
      xlds[buf][r * XSTRIDE + c] = xbase[(__SIZE_TYPE__)r * GRU_T + t0 + c];
    }
#endif

    for (int tl = 0; tl < TCHUNK; ++tl) {
      // x_t for this wave's batch row `col` (lanes 16-31 read a duplicate; masked).
      float xv = xrow[tl];
      ax[0] = (hi == 0) ? (_Float16)xv : (_Float16)0.0f;

      // gates = h @ w_hh^T (+ x,bias folded in).  12x v_wmma_f32_16x16x32_f16.
      v8f aR0 = WMMA(ax, wib[0], WMMA(hA, whh[0], zero));
      v8f aR1 = WMMA(ax, wib[1], WMMA(hA, whh[1], zero));
      v8f aZ0 = WMMA(ax, wib[2], WMMA(hA, whh[2], zero));
      v8f aZ1 = WMMA(ax, wib[3], WMMA(hA, whh[3], zero));
      v8f hN0 = WMMA(hA, whh[4], zero);      // recurrent part of n-gate
      v8f hN1 = WMMA(hA, whh[5], zero);
      v8f iN0 = WMMA(ax, wib[4], zero);      // input part of n-gate (+b_in)
      v8f iN1 = WMMA(ax, wib[5], zero);

      // Elementwise gates in D-layout; lane holds rows m=8*hi+v at column col.
#pragma unroll
      for (int v = 0; v < 8; ++v) {
        float r0 = fsig(aR0[v]);
        float z0 = fsig(aZ0[v]);
        float n0 = ftanh(iN0[v] + r0 * (hN0[v] + bh0));
        float h0 = hD0[v];
        h0 = n0 + z0 * (h0 - n0);
        hD0[v] = h0;
        hlds[(8 * hi + v) * HSTRIDE + col] = (_Float16)h0;

        float r1 = fsig(aR1[v]);
        float z1 = fsig(aZ1[v]);
        float n1 = ftanh(iN1[v] + r1 * (hN1[v] + bh1));
        float h1 = hD1[v];
        h1 = n1 + z1 * (h1 - n1);
        hD1[v] = h1;
        hlds[(8 * hi + v) * HSTRIDE + 16 + col] = (_Float16)h1;
      }

      // Re-load h in A-layout (LDS is in-order per wave; single wave per WG).
      v8h lo = *(const v8h*)&hlds[col * HSTRIDE + 16 * hi];
      v8h hh = *(const v8h*)&hlds[col * HSTRIDE + 16 * hi + 8];
      hA = __builtin_shufflevector(lo, hh, 0, 1, 2, 3, 4, 5, 6, 7,
                                           8, 9, 10, 11, 12, 13, 14, 15);
    }
  }

  // ---- Final FC: out[b] = h_T[b] . fc_w + fc_b (reduce over 16 lanes/half).
  const float fw0 = fc_w[col];
  const float fw1 = fc_w[16 + col];
  const float fb  = fc_b[0];
#pragma unroll
  for (int v = 0; v < 8; ++v) {
    float val = hD0[v] * fw0 + hD1[v] * fw1;
#pragma unroll
    for (int off = 1; off < 16; off <<= 1)
      val += __shfl_xor(val, off, 32);
    if (col == 0) out[bbase + 8 * hi + v] = val + fb;
  }
}

extern "C" void kernel_launch(void* const* d_in, const int* in_sizes, int n_in,
                              void* d_out, int out_size, void* d_ws, size_t ws_size,
                              hipStream_t stream) {
  (void)in_sizes; (void)n_in; (void)out_size; (void)d_ws; (void)ws_size;
  const float* x    = (const float*)d_in[0];
  const float* w_ih = (const float*)d_in[1];
  const float* w_hh = (const float*)d_in[2];
  const float* b_ih = (const float*)d_in[3];
  const float* b_hh = (const float*)d_in[4];
  const float* fc_w = (const float*)d_in[5];
  const float* fc_b = (const float*)d_in[6];
  float* out = (float*)d_out;

  dim3 grid(GRU_B / 16);   // 256 waves, 16 sequences each
  dim3 block(32);          // one wave32 per workgroup
  gru_wmma_kernel<<<grid, block, 0, stream>>>(x, w_ih, w_hh, b_ih, b_hh,
                                              fc_w, fc_b, out);
}
